// ConditionalAttention_24824910970959
// MI455X (gfx1250) — matrix operand order repaired and checked by reading, not compiled, not run on hardware
//
#include <hip/hip_runtime.h>
#include <hip/hip_bf16.h>

typedef __attribute__((ext_vector_type(16))) _Float16 v16h;
typedef __attribute__((ext_vector_type(8)))  float    v8f;

// ---------------------------------------------------------------------------
// Generic row-major GEMM:  C[rows,OUT] = A[rows,K] @ W[OUT,K]^T + bias
// f32 in/out, f16 WMMA compute (v_wmma_f32_16x16x32_f16), f32 accumulate.
// Block = 128 threads = 4 waves; each wave owns a 16-row M-tile.
// - W staged to LDS as f16 once per block (B frags = one contiguous v16h load)
// - A frags loaded as float4 (fragment layout is lane-contiguous in K)
// - Output staged per-wave in LDS, stored as coalesced float4s
// A/B K order is contiguous (k = 16*hi + s, same for both); C layout per CDNA5 ISA 7.12.2 (wave32).
// ---------------------------------------------------------------------------
template<int K, int OUT, bool RELU>
__global__ __launch_bounds__(128)
void wmma_gemm_kernel(const float* __restrict__ A, const float* __restrict__ W,
                      const float* __restrict__ bias, float* __restrict__ C, int rows)
{
    constexpr int KT = K / 32;   // k-steps
    constexpr int NT = OUT / 16; // n-tiles
    __shared__ __align__(32) _Float16 lw[OUT * K];
    __shared__ __align__(16) float    lout[4][16 * OUT];

    for (int i = threadIdx.x; i < OUT * K; i += 128) lw[i] = (_Float16)W[i];
    __syncthreads();

    const int wave    = threadIdx.x >> 5;
    const int lane    = threadIdx.x & 31;
    const int rowBase = blockIdx.x * 64 + wave * 16;
    const int m       = lane & 15;
    const int hi      = lane >> 4;                 // lane half
    int arow = rowBase + m;
    if (arow >= rows) arow = rows - 1;             // clamp (keeps EXEC full for WMMA)
    const float* __restrict__ arp = A + (size_t)arow * K;

    // ---- A fragments: 16x32 f16.  Per lane, elements 0..15 = K[hi*16 .. hi*16+15]
    //      -> contiguous float4 loads.
    v16h afrag[KT];
#pragma unroll
    for (int kt = 0; kt < KT; ++kt) {
        const float4 f0 = *(const float4*)(arp + kt * 32 + hi * 16);
        const float4 f1 = *(const float4*)(arp + kt * 32 + hi * 16 + 4);
        const float4 f2 = *(const float4*)(arp + kt * 32 + hi * 16 + 8);
        const float4 f3 = *(const float4*)(arp + kt * 32 + hi * 16 + 12);
        afrag[kt][0]  = (_Float16)f0.x; afrag[kt][1]  = (_Float16)f0.y;
        afrag[kt][2]  = (_Float16)f0.z; afrag[kt][3]  = (_Float16)f0.w;
        afrag[kt][4]  = (_Float16)f1.x; afrag[kt][5]  = (_Float16)f1.y;
        afrag[kt][6]  = (_Float16)f1.z; afrag[kt][7]  = (_Float16)f1.w;
        afrag[kt][8]  = (_Float16)f2.x; afrag[kt][9]  = (_Float16)f2.y;
        afrag[kt][10] = (_Float16)f2.z; afrag[kt][11] = (_Float16)f2.w;
        afrag[kt][12] = (_Float16)f3.x; afrag[kt][13] = (_Float16)f3.y;
        afrag[kt][14] = (_Float16)f3.z; afrag[kt][15] = (_Float16)f3.w;
    }

    const int n = lane & 15;
    float* __restrict__ lo = lout[wave];
#pragma unroll
    for (int nt = 0; nt < NT; ++nt) {
        v8f acc = {};
#pragma unroll
        for (int kt = 0; kt < KT; ++kt) {
            // B fragment 32x16 f16: per lane, contiguous K[hi*16 .. hi*16+15]
            const v16h b = *(const v16h*)(&lw[(nt * 16 + n) * K + kt * 32 + hi * 16]);
            acc = __builtin_amdgcn_wmma_f32_16x16x32_f16(false, afrag[kt], false, b,
                                                         (short)0, acc, false, false);
        }
        const float bv = bias[nt * 16 + n];
#pragma unroll
        for (int v = 0; v < 8; ++v) {          // C layout: VGPR v -> M = v + hi*8
            float val = acc[v] + bv;
            if (RELU) val = fmaxf(val, 0.f);
            lo[(v + hi * 8) * OUT + nt * 16 + n] = val;   // same-wave LDS, HW in-order
        }
    }

    // ---- coalesced store of the 16 x OUT tile as float4s --------------------
    constexpr int NV4 = (16 * OUT) / (32 * 4);   // float4s per lane
    if (rowBase + 16 <= rows) {                  // full tile: no predication
#pragma unroll
        for (int i = 0; i < NV4; ++i) {
            const int idx = (lane + 32 * i) * 4;
            const int r = idx / OUT, c = idx % OUT;
            *(float4*)(C + (size_t)(rowBase + r) * OUT + c) = *(const float4*)(lo + idx);
        }
    } else {                                     // tail block only
#pragma unroll
        for (int i = 0; i < NV4; ++i) {
            const int idx = (lane + 32 * i) * 4;
            const int r = idx / OUT, c = idx % OUT;
            if (rowBase + r < rows)
                *(float4*)(C + (size_t)(rowBase + r) * OUT + c) = *(const float4*)(lo + idx);
        }
    }
}

// ---------------------------------------------------------------------------
__global__ __launch_bounds__(256)
void zero_kernel(float* __restrict__ p, long long n)
{
    long long i = (long long)blockIdx.x * 256 + threadIdx.x;
    long long stride = (long long)gridDim.x * 256;
    for (; i < n; i += stride) p[i] = 0.f;
}

// ---------------------------------------------------------------------------
// Per-edge: conn_pre = relu(sign(c1)*sqrt(|c1|) + Eb), c1 = (Qdst+Ksrc)*Ew,
// head scores vs Aw, clamp(+-5), exp, atomic segment-sum into ssum[dst].
// One wave per edge, 2 columns per lane.
// ---------------------------------------------------------------------------
__global__ __launch_bounds__(256)
void edge_conn_kernel(const float* __restrict__ qkv, const float* __restrict__ eh,
                      const int* __restrict__ dst, const int* __restrict__ src,
                      const float* __restrict__ Aw,
                      float* __restrict__ conn_pre, float* __restrict__ sc,
                      float* __restrict__ ssum, int E)
{
    const int e = blockIdx.x * 8 + (threadIdx.x >> 5);
    if (e >= E) return;
    const int lane = threadIdx.x & 31;
    const int d = dst[e], s = src[e];

    float partial[2];
#pragma unroll
    for (int i = 0; i < 2; ++i) {
        const int c  = lane + 32 * i;
        const float q  = qkv[(size_t)d * 192 + c];
        const float k  = qkv[(size_t)s * 192 + 64 + c];
        const float ew = eh[(size_t)e * 128 + c];
        const float eb = eh[(size_t)e * 128 + 64 + c];
        const float c1 = (q + k) * ew;
        const float c2 = copysignf(sqrtf(fabsf(c1)), c1);
        const float cv = fmaxf(c2 + eb, 0.f);
        conn_pre[(size_t)e * 64 + c] = cv;
        partial[i] = cv * Aw[(c & 15) * 4 + (c >> 4)];   // Aw[d, h]
    }
    // reduce within the two 16-lane groups (each group = one head per value)
#pragma unroll
    for (int msk = 1; msk <= 8; msk <<= 1) {
        partial[0] += __shfl_xor(partial[0], msk, 32);
        partial[1] += __shfl_xor(partial[1], msk, 32);
    }
    if ((lane & 15) == 0) {
        const int h0 = lane >> 4;                 // heads {0,1}; partial[1] -> {2,3}
        const float s0 = fminf(fmaxf(partial[0], -5.f), 5.f);
        const float s1 = fminf(fmaxf(partial[1], -5.f), 5.f);
        const float e0 = __expf(s0);              // clamp => stable w/o segment_max
        const float e1 = __expf(s1);
        sc[(size_t)e * 4 + h0]     = e0;
        sc[(size_t)e * 4 + 2 + h0] = e1;
        atomicAdd(&ssum[(size_t)d * 4 + h0],     e0);
        atomicAdd(&ssum[(size_t)d * 4 + 2 + h0], e1);
    }
}

// ---------------------------------------------------------------------------
// Per-edge: agg[dst] += (Vsrc + conn_lin) * softmax_weight
// ---------------------------------------------------------------------------
__global__ __launch_bounds__(256)
void msg_agg_kernel(const float* __restrict__ qkv, const float* __restrict__ conn_lin,
                    const float* __restrict__ sc, const float* __restrict__ ssum,
                    const int* __restrict__ dst, const int* __restrict__ src,
                    float* __restrict__ agg, int E)
{
    const int e = blockIdx.x * 8 + (threadIdx.x >> 5);
    if (e >= E) return;
    const int lane = threadIdx.x & 31;
    const int d = dst[e], s = src[e];
#pragma unroll
    for (int i = 0; i < 2; ++i) {
        const int c = lane + 32 * i;
        const int h = c >> 4;
        const float p   = sc[(size_t)e * 4 + h] / (ssum[(size_t)d * 4 + h] + 1e-16f);
        const float val = (qkv[(size_t)s * 192 + 128 + c] + conn_lin[(size_t)e * 64 + c]) * p;
        atomicAdd(&agg[(size_t)d * 64 + c], val);
    }
}

// ---------------------------------------------------------------------------
// LayerNorm over 64 cols, wave per row:  out = LN(in1 [+ in2]) [relu],
// optionally stores the pre-LN sum (residual) to sumout.
// ---------------------------------------------------------------------------
__global__ __launch_bounds__(256)
void ln_kernel(const float* __restrict__ in1, const float* __restrict__ in2,
               const float* __restrict__ g, const float* __restrict__ b,
               float* __restrict__ out, float* __restrict__ sumout,
               int rows, int relu)
{
    const int r = blockIdx.x * 8 + (threadIdx.x >> 5);
    if (r >= rows) return;
    const int lane = threadIdx.x & 31;
    const size_t base = (size_t)r * 64;
    float v0 = in1[base + lane], v1 = in1[base + lane + 32];
    if (in2) { v0 += in2[base + lane]; v1 += in2[base + lane + 32]; }
    if (sumout) { sumout[base + lane] = v0; sumout[base + lane + 32] = v1; }
    float s = v0 + v1;
#pragma unroll
    for (int msk = 1; msk <= 16; msk <<= 1) s += __shfl_xor(s, msk, 32);
    const float mu = s * (1.f / 64.f);
    const float d0 = v0 - mu, d1 = v1 - mu;
    float vs = d0 * d0 + d1 * d1;
#pragma unroll
    for (int msk = 1; msk <= 16; msk <<= 1) vs += __shfl_xor(vs, msk, 32);
    const float inv = rsqrtf(vs * (1.f / 64.f) + 1e-5f);
    float o0 = d0 * inv * g[lane]      + b[lane];
    float o1 = d1 * inv * g[lane + 32] + b[lane + 32];
    if (relu) { o0 = fmaxf(o0, 0.f); o1 = fmaxf(o1, 0.f); }
    out[base + lane] = o0;
    out[base + lane + 32] = o1;
}

// ---------------------------------------------------------------------------
extern "C" void kernel_launch(void* const* d_in, const int* in_sizes, int n_in,
                              void* d_out, int out_size, void* d_ws, size_t ws_size,
                              hipStream_t stream)
{
    const float* x       = (const float*)d_in[0];
    const float* e       = (const float*)d_in[1];
    const int*   dst     = (const int*)  d_in[2];
    const int*   src     = (const int*)  d_in[3];
    const float* qkv_w   = (const float*)d_in[4];
    const float* qkv_b   = (const float*)d_in[5];
    const float* E_w     = (const float*)d_in[6];
    const float* E_b     = (const float*)d_in[7];
    const float* Aw      = (const float*)d_in[8];
    const float* conn1_w = (const float*)d_in[9];
    const float* conn1_b = (const float*)d_in[10];
    const float* conn2_w = (const float*)d_in[11];
    const float* conn2_b = (const float*)d_in[12];
    const float* ffn1_w  = (const float*)d_in[13];
    const float* ffn1_b  = (const float*)d_in[14];
    const float* ffn2_w  = (const float*)d_in[15];
    const float* ffn2_b  = (const float*)d_in[16];
    const float* ln1h_g  = (const float*)d_in[17];
    const float* ln1h_b  = (const float*)d_in[18];
    const float* ln2h_g  = (const float*)d_in[19];
    const float* ln2h_b  = (const float*)d_in[20];
    const float* ln1c_g  = (const float*)d_in[21];
    const float* ln1c_b  = (const float*)d_in[22];
    const float* ln2c_g  = (const float*)d_in[23];
    const float* ln2c_b  = (const float*)d_in[24];

    const int N = in_sizes[0] / 64;   // 50000
    const int E = in_sizes[2];        // 512000

    // ---- workspace carve-up (floats) -------------------------------------
    float* ws = (float*)d_ws;
    size_t o = 0;
    float* qkv      = ws + o; o += (size_t)N * 192;
    float* eh       = ws + o; o += (size_t)E * 128;   // reused: conn_lin, conn2out
    float* conn_pre = ws + o; o += (size_t)E * 64;    // reused: cn
    float* sc       = ws + o; o += (size_t)E * 4;
    float* ssum     = ws + o; o += (size_t)N * 4;     // ssum+agg contiguous -> one zero pass
    float* agg      = ws + o; o += (size_t)N * 64;
    float* hres     = ws + o; o += (size_t)N * 64;
    float* hn       = ws + o; o += (size_t)N * 64;
    float* ffn1o    = ws + o; o += (size_t)N * 128;
    float* ffn2o    = ws + o; o += (size_t)N * 64;
    float* conn_lin = eh;            // eh dead after edge_conn_kernel
    float* cn       = conn_pre;      // conn_pre dead after conn1 GEMM
    float* conn2o   = eh;            // conn_lin dead after edge_post1

    float* out_h    = (float*)d_out;            // [N,64]
    float* out_conn = out_h + (size_t)N * 64;   // [E,64]

    const int gN  = (N + 63) / 64;   // GEMM blocks over nodes
    const int gE  = (E + 63) / 64;   // GEMM blocks over edges
    const int eW  = (E + 7) / 8;     // wave-per-edge blocks
    const int lnN = (N + 7) / 8;
    const int lnE = (E + 7) / 8;

    // 0) zero the atomic accumulators (ssum + agg, contiguous)
    zero_kernel<<<512, 256, 0, stream>>>(ssum, (long long)N * 68);
    // 1) qkv = x @ qkv_w^T + b            [N,192]
    wmma_gemm_kernel<64, 192, false><<<gN, 128, 0, stream>>>(x, qkv_w, qkv_b, qkv, N);
    // 2) Eh = e @ E_w^T + b               [E,128]  (Ew | Eb)
    wmma_gemm_kernel<64, 128, false><<<gE, 128, 0, stream>>>(e, E_w, E_b, eh, E);
    // 3) per-edge conn_pre + exp(score) + segment-sum
    edge_conn_kernel<<<eW, 256, 0, stream>>>(qkv, eh, dst, src, Aw, conn_pre, sc, ssum, E);
    // 4) conn_lin = conn_pre @ conn1_w^T + b   [E,64]
    wmma_gemm_kernel<64, 64, false><<<gE, 128, 0, stream>>>(conn_pre, conn1_w, conn1_b, conn_lin, E);
    // 5) agg[dst] += (Vsrc + conn_lin) * softmax
    msg_agg_kernel<<<eW, 256, 0, stream>>>(qkv, conn_lin, sc, ssum, dst, src, agg, E);
    // 6) node path: hres = x + agg; hn = LN1h(hres)
    ln_kernel<<<lnN, 256, 0, stream>>>(x, agg, ln1h_g, ln1h_b, hn, hres, N, 0);
    // 7) FFN
    wmma_gemm_kernel<64, 128, true ><<<gN, 128, 0, stream>>>(hn, ffn1_w, ffn1_b, ffn1o, N);
    wmma_gemm_kernel<128, 64, false><<<gN, 128, 0, stream>>>(ffn1o, ffn2_w, ffn2_b, ffn2o, N);
    // 8) out_h = LN2h(ffn2o + hres)
    ln_kernel<<<lnN, 256, 0, stream>>>(ffn2o, hres, ln2h_g, ln2h_b, out_h, nullptr, N, 0);
    // 9) edge path: cn = relu(LN1c(conn_lin))
    ln_kernel<<<lnE, 256, 0, stream>>>(conn_lin, nullptr, ln1c_g, ln1c_b, cn, nullptr, E, 1);
    // 10) conn2o = cn @ conn2_w^T + b
    wmma_gemm_kernel<64, 64, false><<<gE, 128, 0, stream>>>(cn, conn2_w, conn2_b, conn2o, E);
    // 11) out_conn = LN2c(conn2o + e)
    ln_kernel<<<lnE, 256, 0, stream>>>(conn2o, e, ln2c_g, ln2c_b, out_conn, nullptr, E, 0);

    (void)n_in; (void)out_size; (void)ws_size;
}
